// Block_15418932593387
// MI455X (gfx1250) — compile-verified, matrix operand order checked by reading, not run on hardware
//
#include <hip/hip_runtime.h>

// ---------------------------------------------------------------------------
// Types for CDNA5 WMMA (wave32, v_wmma_f32_16x16x32_bf16)
// ---------------------------------------------------------------------------
typedef __bf16 bf16;
typedef __attribute__((ext_vector_type(16))) __bf16 v16bf;
typedef __attribute__((ext_vector_type(8)))  __bf16 bf16x8;
typedef __attribute__((ext_vector_type(8)))  float  v8f;

static __device__ inline v8f wmma_bf16(v16bf a, v16bf b, v8f c) {
  // (neg_a, A, neg_b, B, c_mod, C, reuse_a, reuse_b)
  return __builtin_amdgcn_wmma_f32_16x16x32_bf16(false, a, false, b, (short)0, c,
                                                 false, false);
}

// Build a 16-element bf16 fragment from two contiguous 16-byte chunks.
// Per the ISA 16-bit operand layout, each lane needs K-chunk [half*8 .. +7]
// and [16+half*8 .. +7]; callers pass the two 16B-aligned pointers.
static __device__ inline v16bf make_frag(const bf16* p0, const bf16* p1) {
  bf16x8 lo = *(const bf16x8*)p0;
  bf16x8 hi = *(const bf16x8*)p1;
  return __builtin_shufflevector(lo, hi, 0, 1, 2, 3, 4, 5, 6, 7,
                                 8, 9, 10, 11, 12, 13, 14, 15);
}

// gfx1250 async copy: 16B global -> LDS, tracked by ASYNCcnt (ISA 15.18.3 op 98)
static __device__ inline void async_load_b128(const bf16* gptr, bf16* lptr) {
  unsigned long long ga = (unsigned long long)(uintptr_t)gptr;
  unsigned la = (unsigned)(uintptr_t)lptr;  // low 32 bits = LDS byte address
  asm volatile("global_load_async_to_lds_b128 %0, %1, off"
               :
               : "v"(la), "v"(ga)
               : "memory");
}
// ASYNCcnt completes in order: allow the N newest loads to stay in flight.
static __device__ inline void wait_async_le8() {
  asm volatile("s_wait_asynccnt 0x8" ::: "memory");
}

// ---------------------------------------------------------------------------
// Weight prep: fp32 -> bf16, K-contiguous (transposed) layouts
// ---------------------------------------------------------------------------
// Wq/Wk/Wv: [H, D, 64] fp32  ->  Wt[n*D + d] bf16, n = h*64+e
__global__ __launch_bounds__(256) void pack_qkv_kernel(const float* __restrict__ W,
                                                       bf16* __restrict__ Wt, int D) {
  size_t idx = (size_t)blockIdx.x * 256 + threadIdx.x;
  if (idx >= (size_t)D * D) return;
  int n = (int)(idx / D), d = (int)(idx % D);
  int h = n >> 6, e = n & 63;
  Wt[idx] = (bf16)W[((size_t)h * D + d) * 64 + e];
}

// W [K,N] fp32 row-major -> Wt[n*K + k] bf16
__global__ __launch_bounds__(256) void transpose_cvt_kernel(const float* __restrict__ W,
                                                            bf16* __restrict__ Wt,
                                                            int K, int N) {
  size_t idx = (size_t)blockIdx.x * 256 + threadIdx.x;
  if (idx >= (size_t)K * N) return;
  size_t n = idx / K, k = idx % K;
  Wt[idx] = (bf16)W[k * (size_t)N + n];
}

// ---------------------------------------------------------------------------
// LayerNorm over D per row, bf16 output
// ---------------------------------------------------------------------------
__global__ __launch_bounds__(256) void ln_kernel(const float* __restrict__ x,
                                                 const float* __restrict__ g,
                                                 const float* __restrict__ be,
                                                 bf16* __restrict__ out, int D) {
  int row = blockIdx.x;
  const float* xr = x + (size_t)row * D;
  float s = 0.f, s2 = 0.f;
  for (int i = threadIdx.x; i < D; i += 256) {
    float v = xr[i];
    s += v;
    s2 += v * v;
  }
  __shared__ float red[2][8];
  for (int off = 16; off; off >>= 1) {
    s += __shfl_down(s, off, 32);
    s2 += __shfl_down(s2, off, 32);
  }
  int wid = threadIdx.x >> 5;
  if ((threadIdx.x & 31) == 0) { red[0][wid] = s; red[1][wid] = s2; }
  __syncthreads();
  if (threadIdx.x == 0) {
    float a = 0.f, b = 0.f;
    for (int i = 0; i < 8; ++i) { a += red[0][i]; b += red[1][i]; }
    red[0][0] = a;
    red[1][0] = b;
  }
  __syncthreads();
  float mean = red[0][0] / (float)D;
  float var = red[1][0] / (float)D - mean * mean;
  float rstd = rsqrtf(var + 1e-5f);
  for (int i = threadIdx.x; i < D; i += 256) {
    float v = (xr[i] - mean) * rstd * g[i] + be[i];
    out[(size_t)row * D + i] = (bf16)v;
  }
}

// ---------------------------------------------------------------------------
// Tiled WMMA GEMM: C[M,N] = A[M,K](bf16,row-major) x Bt[N,K](bf16,K-contig)
// Block tile 128x128, BK=64 (16 WMMA / barrier pair), 8 waves (2 x 4),
// each wave 64x32 (4x2 WMMA tiles).
// Staging: double-buffered global_load_async_to_lds_b128; next tile's loads
// are issued before s_wait_asynccnt<=8 so HBM fetch overlaps the WMMAs.
// Epilogue modes:
//   0: bf16 head layout   dst[((b*H+h)*T + t)*64 + e]        (Q, K)
//   1: bf16 V-transposed  dst[((b*H+h)*64 + e)*T + t]        (V)
//   2: bf16 row-major, +bias, ReLU                           (FFN1)
//   3: f32  row-major, +bias, +residual                      (Wo out, FFN2 out)
// ---------------------------------------------------------------------------
#define BM 128
#define BN 128
#define BKG 64
#define LDPG 80   // padded LDS pitch in bf16 (160B rows; every 16B chunk aligned)
#define LDPA 40   // attention P-tile pitch

__global__ __launch_bounds__(256) void gemm_bf16_kernel(
    const bf16* __restrict__ A, const bf16* __restrict__ Bt, int M, int N, int K,
    int mode, const float* __restrict__ bias, const float* __restrict__ resid,
    void* __restrict__ dstv, int Hh, int Tt) {
  __shared__ __align__(16) bf16 As[2][BM * LDPG];
  __shared__ __align__(16) bf16 Bs[2][BN * LDPG];

  const int tid = threadIdx.x;
  const int lane = tid & 31;
  const int wave = tid >> 5;
  const int wm = wave >> 2;  // 0..1
  const int wn = wave & 3;   // 0..3
  const int half = lane >> 4;
  const int l15 = lane & 15;
  const int bMr = blockIdx.x * BM;
  const int bNr = blockIdx.y * BN;

  // Per-thread staging addresses: 4 16B chunks per tile per matrix.
  const int chunk0 = tid * 4;
  const int srow = chunk0 >> 3;          // 0..127 (same row for all 4 chunks)
  (void)srow;

  auto issue_tile = [&](int k0, int buf) {
#pragma unroll
    for (int c = 0; c < 4; ++c) {
      int chunk = chunk0 + c;        // 0..1023
      int row = chunk >> 3;          // 0..127
      int off = (chunk & 7) * 8;     // bf16 elems (16B chunks)
      async_load_b128(&A[(size_t)(bMr + row) * K + k0 + off],
                      &As[buf][row * LDPG + off]);
      async_load_b128(&Bt[(size_t)(bNr + row) * K + k0 + off],
                      &Bs[buf][row * LDPG + off]);
    }
  };

  v8f acc[4][2];
#pragma unroll
  for (int mi = 0; mi < 4; ++mi)
#pragma unroll
    for (int ni = 0; ni < 2; ++ni)
      acc[mi][ni] = (v8f){0.f, 0.f, 0.f, 0.f, 0.f, 0.f, 0.f, 0.f};

  issue_tile(0, 0);
  int cur = 0;
  for (int k0 = 0; k0 < K; k0 += BKG) {
    // Prefetch next tile into the other buffer (clamped redundant load on the
    // final iteration keeps the loop branch-free; that buffer is dead anyway).
    int knext = (k0 + BKG < K) ? (k0 + BKG) : k0;
    issue_tile(knext, cur ^ 1);
    wait_async_le8();   // oldest 8 (current tile) complete; newest 8 in flight
    __syncthreads();

#pragma unroll
    for (int ks = 0; ks < 2; ++ks) {
      const int kb = ks * 32;
      v16bf af[4], bfr[2];
#pragma unroll
      for (int mi = 0; mi < 4; ++mi) {
        int r = wm * 64 + mi * 16 + l15;
        af[mi] = make_frag(&As[cur][r * LDPG + kb + half * 8],
                           &As[cur][r * LDPG + kb + 16 + half * 8]);
      }
#pragma unroll
      for (int ni = 0; ni < 2; ++ni) {
        int n = wn * 32 + ni * 16 + l15;
        bfr[ni] = make_frag(&Bs[cur][n * LDPG + kb + half * 8],
                            &Bs[cur][n * LDPG + kb + 16 + half * 8]);
      }
#pragma unroll
      for (int mi = 0; mi < 4; ++mi)
#pragma unroll
        for (int ni = 0; ni < 2; ++ni)
          acc[mi][ni] = wmma_bf16(af[mi], bfr[ni], acc[mi][ni]);
    }
    __syncthreads();   // all waves done reading before this buffer is reloaded
    cur ^= 1;
  }

  // Epilogue. C layout: lane holds col n=l15; vec elem i is row i + 8*half.
#pragma unroll
  for (int mi = 0; mi < 4; ++mi) {
#pragma unroll
    for (int ni = 0; ni < 2; ++ni) {
      int cc = bNr + wn * 32 + ni * 16 + l15;
      float bv = bias ? bias[cc] : 0.0f;
#pragma unroll
      for (int i = 0; i < 8; ++i) {
        int rr = bMr + wm * 64 + mi * 16 + i + half * 8;
        float v = acc[mi][ni][i];
        if (mode == 0) {
          int b = rr / Tt, t = rr % Tt;
          int h = cc >> 6, e = cc & 63;
          ((bf16*)dstv)[(((size_t)(b * Hh + h)) * Tt + t) * 64 + e] = (bf16)v;
        } else if (mode == 1) {
          int b = rr / Tt, t = rr % Tt;
          int h = cc >> 6, e = cc & 63;
          ((bf16*)dstv)[(((size_t)(b * Hh + h)) * 64 + e) * Tt + t] = (bf16)v;
        } else if (mode == 2) {
          ((bf16*)dstv)[(size_t)rr * N + cc] = (bf16)fmaxf(v + bv, 0.0f);
        } else {
          ((float*)dstv)[(size_t)rr * N + cc] = v + bv + resid[(size_t)rr * N + cc];
        }
      }
    }
  }
}

// ---------------------------------------------------------------------------
// Causal flash attention: one wave per 16-row q tile; HS=64.
// q,k: [B*H, T, 64] bf16 ;  vt: [B*H, 64, T] bf16 ;  out: [B*T, H*64] bf16
// ---------------------------------------------------------------------------
__global__ __launch_bounds__(256) void attn_kernel(const bf16* __restrict__ q,
                                                   const bf16* __restrict__ k,
                                                   const bf16* __restrict__ vt,
                                                   bf16* __restrict__ out, int H,
                                                   int T) {
  __shared__ __align__(16) bf16 P[8][16 * LDPA];  // per-wave 16 x 32 (padded)
  const int lane = threadIdx.x & 31;
  const int wave = threadIdx.x >> 5;
  const int half = lane >> 4;
  const int l15 = lane & 15;

  const int tilesPerSeq = T >> 4;
  const int tile = blockIdx.x * 8 + wave;
  const int bh = tile / tilesPerSeq;
  const int t0 = (tile % tilesPerSeq) * 16;
  const int b = bh / H, h = bh % H;

  const bf16* qb = q + (size_t)bh * T * 64;
  const bf16* kb = k + (size_t)bh * T * 64;
  const bf16* vb = vt + (size_t)bh * 64 * T;
  bf16* Pw = &P[wave][0];

  // q fragments (K = e: 0..31 and 32..63)
  const bf16* qrow = qb + (size_t)(t0 + l15) * 64;
  v16bf qf0 = make_frag(qrow + half * 8, qrow + 16 + half * 8);
  v16bf qf1 = make_frag(qrow + 32 + half * 8, qrow + 48 + half * 8);

  float m_i[8], l_i[8];
  v8f o_acc[4];
#pragma unroll
  for (int i = 0; i < 8; ++i) { m_i[i] = -3.0e38f; l_i[i] = 0.f; }
#pragma unroll
  for (int j = 0; j < 4; ++j)
    o_acc[j] = (v8f){0.f, 0.f, 0.f, 0.f, 0.f, 0.f, 0.f, 0.f};

  const float scale = 0.125f;           // 1/sqrt(64)
  const float log2e = 1.44269504088896f;

  for (int s0 = 0; s0 <= t0 + 15; s0 += 32) {
    // ---- scores: two 16x16 tiles over s columns ----
    v8f S[2];
#pragma unroll
    for (int st = 0; st < 2; ++st) {
      const bf16* krow = kb + (size_t)(s0 + st * 16 + l15) * 64;
      v16bf kf0 = make_frag(krow + half * 8, krow + 16 + half * 8);
      v16bf kf1 = make_frag(krow + 32 + half * 8, krow + 48 + half * 8);
      v8f sa = (v8f){0.f, 0.f, 0.f, 0.f, 0.f, 0.f, 0.f, 0.f};
      sa = wmma_bf16(qf0, kf0, sa);
      sa = wmma_bf16(qf1, kf1, sa);
      S[st] = sa;
    }
    // ---- scale + causal mask ----
#pragma unroll
    for (int st = 0; st < 2; ++st) {
      int sc = s0 + st * 16 + l15;
#pragma unroll
      for (int i = 0; i < 8; ++i) {
        int row = t0 + i + half * 8;
        float v = S[st][i] * scale;
        S[st][i] = (sc <= row) ? v : -3.0e38f;
      }
    }
    // ---- online softmax (row reductions across 16-lane half-groups) ----
#pragma unroll
    for (int i = 0; i < 8; ++i) {
      float mx = fmaxf(S[0][i], S[1][i]);
      for (int off = 8; off; off >>= 1) mx = fmaxf(mx, __shfl_xor(mx, off, 16));
      float mnew = fmaxf(m_i[i], mx);
      float alpha = exp2f((m_i[i] - mnew) * log2e);
      float p0 = exp2f((S[0][i] - mnew) * log2e);
      float p1 = exp2f((S[1][i] - mnew) * log2e);
      float rs = p0 + p1;
      for (int off = 8; off; off >>= 1) rs += __shfl_xor(rs, off, 16);
      l_i[i] = l_i[i] * alpha + rs;
      m_i[i] = mnew;
#pragma unroll
      for (int j = 0; j < 4; ++j) o_acc[j][i] *= alpha;
      int prow = i + half * 8;  // C layout -> LDS row-major 16x32
      Pw[prow * LDPA + l15] = (bf16)p0;
      Pw[prow * LDPA + 16 + l15] = (bf16)p1;
    }
    // ---- P (16x32, A-frag via LDS) x Vt (32x64) ----
    const bf16* pr = &Pw[l15 * LDPA];
    v16bf pf = make_frag(pr + half * 8, pr + 16 + half * 8);
#pragma unroll
    for (int j = 0; j < 4; ++j) {
      const bf16* vrow = vb + (size_t)(j * 16 + l15) * T + s0;
      v16bf vf = make_frag(vrow + half * 8, vrow + 16 + half * 8);
      o_acc[j] = wmma_bf16(pf, vf, o_acc[j]);
    }
  }

  // ---- finalize: o / l, heads concatenated along feature dim ----
#pragma unroll
  for (int i = 0; i < 8; ++i) {
    float inv = 1.0f / l_i[i];
    int row = t0 + i + half * 8;
    size_t base = ((size_t)b * T + row) * (size_t)(H * 64) + (size_t)h * 64;
#pragma unroll
    for (int j = 0; j < 4; ++j)
      out[base + j * 16 + l15] = (bf16)(o_acc[j][i] * inv);
  }
}

// ---------------------------------------------------------------------------
// Launch
// ---------------------------------------------------------------------------
extern "C" void kernel_launch(void* const* d_in, const int* in_sizes, int n_in,
                              void* d_out, int out_size, void* d_ws, size_t ws_size,
                              hipStream_t stream) {
  (void)in_sizes; (void)n_in; (void)out_size; (void)ws_size;
  const int Bb = 4, Tt = 2048, Dd = 1024, Hh = 16, FF = 4096;
  const int Mrows = Bb * Tt;  // 8192

  const float* x = (const float*)d_in[0];
  const float* Wq = (const float*)d_in[1];
  const float* Wk = (const float*)d_in[2];
  const float* Wv = (const float*)d_in[3];
  const float* Wo = (const float*)d_in[4];
  const float* bo = (const float*)d_in[5];
  const float* W1 = (const float*)d_in[6];
  const float* b1 = (const float*)d_in[7];
  const float* W2 = (const float*)d_in[8];
  const float* b2 = (const float*)d_in[9];
  const float* g1 = (const float*)d_in[10];
  const float* be1 = (const float*)d_in[11];
  const float* g2 = (const float*)d_in[12];
  const float* be2 = (const float*)d_in[13];

  char* w = (char*)d_ws;
  const size_t MB = 1024ull * 1024ull;
  bf16* hln  = (bf16*)(w + 0 * MB);    // 16 MB  [B*T, D]     (reused as h2 later)
  bf16* Wqt  = (bf16*)(w + 16 * MB);   //  2 MB
  bf16* Wkt  = (bf16*)(w + 18 * MB);   //  2 MB
  bf16* Wvt  = (bf16*)(w + 20 * MB);   //  2 MB
  bf16* Wot  = (bf16*)(w + 22 * MB);   //  2 MB
  bf16* W1t  = (bf16*)(w + 24 * MB);   //  8 MB
  bf16* W2t  = (bf16*)(w + 32 * MB);   //  8 MB
  bf16* qbuf = (bf16*)(w + 40 * MB);   // 16 MB  [B*H, T, 64]
  bf16* kbuf = (bf16*)(w + 56 * MB);   // 16 MB
  bf16* vtb  = (bf16*)(w + 72 * MB);   // 16 MB  [B*H, 64, T]
  bf16* aout = (bf16*)(w + 88 * MB);   // 16 MB  [B*T, D]
  float* x1  = (float*)(w + 104 * MB); // 32 MB  [B*T, D] f32
  bf16* ff1  = (bf16*)(w + 40 * MB);   // 64 MB  reuses q/k/vt/aout region
  bf16* h2   = hln;                    // reuses hln

  // --- weight prep ---
  pack_qkv_kernel<<<(Dd * Dd + 255) / 256, 256, 0, stream>>>(Wq, Wqt, Dd);
  pack_qkv_kernel<<<(Dd * Dd + 255) / 256, 256, 0, stream>>>(Wk, Wkt, Dd);
  pack_qkv_kernel<<<(Dd * Dd + 255) / 256, 256, 0, stream>>>(Wv, Wvt, Dd);
  transpose_cvt_kernel<<<(Dd * Dd + 255) / 256, 256, 0, stream>>>(Wo, Wot, Dd, Dd);
  transpose_cvt_kernel<<<(Dd * FF + 255) / 256, 256, 0, stream>>>(W1, W1t, Dd, FF);
  transpose_cvt_kernel<<<(FF * Dd + 255) / 256, 256, 0, stream>>>(W2, W2t, FF, Dd);

  // --- LN1 ---
  ln_kernel<<<Mrows, 256, 0, stream>>>(x, g1, be1, hln, Dd);

  // --- QKV projections ---
  dim3 gD(Mrows / BM, Dd / BN);
  gemm_bf16_kernel<<<gD, 256, 0, stream>>>(hln, Wqt, Mrows, Dd, Dd, 0, nullptr,
                                           nullptr, qbuf, Hh, Tt);
  gemm_bf16_kernel<<<gD, 256, 0, stream>>>(hln, Wkt, Mrows, Dd, Dd, 0, nullptr,
                                           nullptr, kbuf, Hh, Tt);
  gemm_bf16_kernel<<<gD, 256, 0, stream>>>(hln, Wvt, Mrows, Dd, Dd, 1, nullptr,
                                           nullptr, vtb, Hh, Tt);

  // --- attention ---
  int tiles = Bb * Hh * (Tt / 16);  // 8192 waves
  attn_kernel<<<tiles / 8, 256, 0, stream>>>(qbuf, kbuf, vtb, aout, Hh, Tt);

  // --- output projection + residual ---
  gemm_bf16_kernel<<<gD, 256, 0, stream>>>(aout, Wot, Mrows, Dd, Dd, 3, bo, x, x1,
                                           Hh, Tt);

  // --- LN2 ---
  ln_kernel<<<Mrows, 256, 0, stream>>>(x1, g2, be2, h2, Dd);

  // --- FFN ---
  dim3 gF1(Mrows / BM, FF / BN);
  gemm_bf16_kernel<<<gF1, 256, 0, stream>>>(h2, W1t, Mrows, FF, Dd, 2, b1, nullptr,
                                            ff1, Hh, Tt);
  gemm_bf16_kernel<<<gD, 256, 0, stream>>>(ff1, W2t, Mrows, Dd, FF, 3, b2, x1,
                                           (float*)d_out, Hh, Tt);
}
